// QRNNEncoder_55370718380156
// MI455X (gfx1250) — matrix operand order repaired and checked by reading, not compiled, not run on hardware
//
#include <hip/hip_runtime.h>
#include <hip/hip_bf16.h>
#include <math.h>

// Problem constants (from the reference)
#define S_LEN  256
#define B_SZ   32
#define E_DIM  300
#define H_DIM  2400
#define N3H    7200                 // 3*H
#define M_ROWS (S_LEN * B_SZ)       // 8192 GEMM rows (s*B + b)

// GEMM macro-tile
#define BM 128
#define BN 96
#define BK 32
#define KP_MAX 2432                 // H padded to multiple of 64 -> KT always even
#define LDS_STRIDE 40               // 32 + 8 bf16 pad -> 80B rows, 16B aligned, spreads banks

typedef __attribute__((ext_vector_type(8)))  float  v8f;
typedef __attribute__((ext_vector_type(8)))  __bf16 v8bf;
typedef __attribute__((ext_vector_type(16))) __bf16 v16bf;

static __device__ __forceinline__ v16bf join16(v8bf lo, v8bf hi) {
  v16bf r;
#pragma unroll
  for (int i = 0; i < 8; ++i) { r[i] = lo[i]; r[i + 8] = hi[i]; }
  return r;
}

// Async global -> LDS, 16B per lane, tracked by ASYNCcnt (CDNA5 path).
static __device__ __forceinline__ void async_ld_b128(unsigned lds_addr,
                                                     const void* gaddr) {
  asm volatile("global_load_async_to_lds_b128 %0, %1, off"
               :: "v"(lds_addr), "v"(gaddr)
               : "memory");
}
static __device__ __forceinline__ void wait_async0() {
  asm volatile("s_wait_asynccnt 0x0" ::: "memory");
}

// ---------------------------------------------------------------------------
// fp32 [rows,K] -> bf16 [rows,Kp], zero-padded in K
// ---------------------------------------------------------------------------
__global__ void cvt_pad_bf16(const float* __restrict__ src, __bf16* __restrict__ dst,
                             int rows, int K, int Kp) {
  int i = blockIdx.x * blockDim.x + threadIdx.x;
  int total = rows * Kp;
  if (i >= total) return;
  int r = i / Kp;
  int k = i - r * Kp;
  float v = (k < K) ? src[(size_t)r * K + k] : 0.0f;
  dst[i] = (__bf16)v;
}

__global__ void zero_f32(float* __restrict__ p, int n) {
  int i = blockIdx.x * blockDim.x + threadIdx.x;
  if (i < n) p[i] = 0.0f;
}

// ---------------------------------------------------------------------------
// Gates GEMM:  gates[m, n] = act( sum_k Xb[m,k] * Wb[n,k] + bias[n] )
//   m = s*B + b (8192 rows), n in [0, 7200); act: n/H==0 -> tanh, else sigmoid
//   bf16 A/B, fp32 accumulate via v_wmma_f32_16x16x32_bf16
// Grid: (N3H/BN = 75, M_ROWS/BM = 64), block 256 (8 waves: 4 in M x 2 in N)
// Double-buffered LDS, async global->LDS staging, K-loop unrolled x2
// (KT is always even: 320/32=10, 2432/32=76).
// ---------------------------------------------------------------------------
struct GemmCtx {
  const __bf16* sAb;  // sA[buf] base (as generic pointer)
  const __bf16* sBb;
};

__global__ __launch_bounds__(256) void gemm_gates(
    const __bf16* __restrict__ Xb, const __bf16* __restrict__ Wb,
    const float* __restrict__ bias, float* __restrict__ gates, int Kp) {

  __shared__ __align__(16) __bf16 sA[2][BM * LDS_STRIDE];   // 2 x 128x32 (padded)
  __shared__ __align__(16) __bf16 sB[2][BN * LDS_STRIDE];   // 2 x  96x32 (padded)

  const int t    = threadIdx.x;
  const int lane = t & 31;
  const int wave = t >> 5;
  const int idx  = lane & 15;        // row/col index within 16-wide fragment
  const int half = lane >> 4;        // 0: lanes 0-15, 1: lanes 16-31
  const int wm   = wave & 3;         // 4 waves along M
  const int wn   = wave >> 2;        // 2 waves along N
  const int m0   = blockIdx.y * BM;
  const int n0   = blockIdx.x * BN;

  // ---- per-thread staging assignment (16B chunks of 8 bf16) --------------
  // A: 128x32 = 512 chunks; thread t loads rows t>>2 and 64+(t>>2), col (t&3)*8
  // B:  96x32 = 384 chunks; all threads load row t>>2; threads<128 also 64+(t>>2)
  const int rA0 = t >> 2;
  const int rA1 = rA0 + 64;
  const int cC  = (t & 3) << 3;
  const bool doB1 = (t < 128);

  const __bf16* gA0 = Xb + (size_t)(m0 + rA0) * Kp + cC;
  const __bf16* gA1 = Xb + (size_t)(m0 + rA1) * Kp + cC;
  const __bf16* gB0 = Wb + (size_t)(n0 + rA0) * Kp + cC;
  const __bf16* gB1 = Wb + (size_t)(n0 + rA1) * Kp + cC;   // used only if doB1

  unsigned lA0[2], lA1[2], lB0[2], lB1[2];
#pragma unroll
  for (int buf = 0; buf < 2; ++buf) {
    lA0[buf] = (unsigned)(uintptr_t)&sA[buf][rA0 * LDS_STRIDE + cC];
    lA1[buf] = (unsigned)(uintptr_t)&sA[buf][rA1 * LDS_STRIDE + cC];
    lB0[buf] = (unsigned)(uintptr_t)&sB[buf][rA0 * LDS_STRIDE + cC];
    lB1[buf] = (unsigned)(uintptr_t)&sB[buf][rA1 * LDS_STRIDE + cC];
  }

  v8f acc[2][3];
#pragma unroll
  for (int mi = 0; mi < 2; ++mi)
#pragma unroll
    for (int nj = 0; nj < 3; ++nj) acc[mi][nj] = (v8f)0.0f;

  const int arow0 = (wm * 32 + idx) * LDS_STRIDE;       // + mi*16*LDS_STRIDE
  const int brow0 = (wn * 48 + idx) * LDS_STRIDE;       // + nj*16*LDS_STRIDE

  // One K-step of compute from LDS buffer `buf` (buf is a literal at call site)
  auto compute = [&](const __bf16* __restrict__ a, const __bf16* __restrict__ b) {
    // A (16x32): elems 0..7 -> K=h*8+e ; elems 8..15 -> K=16+h*8+e
    v16bf afrag[2];
#pragma unroll
    for (int mi = 0; mi < 2; ++mi) {
      int off = arow0 + mi * (16 * LDS_STRIDE);
      v8bf lo = *(const v8bf*)&a[off + half * 8];
      v8bf hi = *(const v8bf*)&a[off + 16 + half * 8];
      afrag[mi] = join16(lo, hi);
    }
    // B (32x16): elems 0..7 -> K=h*8+e ; elems 8..15 -> K=16+h*8+e
    v16bf bfrag[3];
#pragma unroll
    for (int nj = 0; nj < 3; ++nj) {
      int off = brow0 + nj * (16 * LDS_STRIDE);
      v8bf lo = *(const v8bf*)&b[off + half * 8];
      v8bf hi = *(const v8bf*)&b[off + 16 + half * 8];
      bfrag[nj] = join16(lo, hi);
    }
#pragma unroll
    for (int mi = 0; mi < 2; ++mi)
#pragma unroll
      for (int nj = 0; nj < 3; ++nj)
        acc[mi][nj] = __builtin_amdgcn_wmma_f32_16x16x32_bf16(
            false, afrag[mi], false, bfrag[nj], (short)0, acc[mi][nj],
            false, false);
  };

  const int KT = Kp / BK;  // even by construction

  // prefetch tile 0 into buffer 0
  async_ld_b128(lA0[0], gA0);
  async_ld_b128(lA1[0], gA1);
  async_ld_b128(lB0[0], gB0);
  if (doB1) async_ld_b128(lB1[0], gB1);

  for (int kt = 0; kt < KT; kt += 2) {
    // ---- even step: consume buf0, prefetch tile kt+1 into buf1 ----------
    wait_async0();
    __syncthreads();
    {
      const int ko = (kt + 1) * BK;              // kt+1 < KT always (KT even)
      async_ld_b128(lA0[1], gA0 + ko);
      async_ld_b128(lA1[1], gA1 + ko);
      async_ld_b128(lB0[1], gB0 + ko);
      if (doB1) async_ld_b128(lB1[1], gB1 + ko);
    }
    compute(sA[0], sB[0]);

    // ---- odd step: consume buf1, prefetch tile kt+2 into buf0 -----------
    wait_async0();
    __syncthreads();
    if (kt + 2 < KT) {
      const int ko = (kt + 2) * BK;
      async_ld_b128(lA0[0], gA0 + ko);
      async_ld_b128(lA1[0], gA1 + ko);
      async_ld_b128(lB0[0], gB0 + ko);
      if (doB1) async_ld_b128(lB1[0], gB1 + ko);
    }
    compute(sA[1], sB[1]);
  }

  // ---- epilogue: bias + activation, store fp32 gates ----------------------
  // D layout: vgpr r -> row = half*8 + r (within 16), col = idx
#pragma unroll
  for (int nj = 0; nj < 3; ++nj) {
    int coln  = n0 + wn * 48 + nj * 16 + idx;
    float bv  = bias[coln];
    int gate  = coln / H_DIM;                // 0=Z(tanh), 1=F(sig), 2=O(sig)
#pragma unroll
    for (int mi = 0; mi < 2; ++mi) {
#pragma unroll
      for (int r = 0; r < 8; ++r) {
        int row  = m0 + wm * 32 + mi * 16 + half * 8 + r;
        float v  = acc[mi][nj][r] + bv;
        v = (gate == 0) ? tanhf(v) : (1.0f / (1.0f + expf(-v)));
        gates[(size_t)row * N3H + coln] = v;
      }
    }
  }
}

// ---------------------------------------------------------------------------
// fo-pool scan: one thread per (b,h); c = f*c + (1-f)*z; x_next = o*c;
// csum[b,h] += c_last (accumulated across the 3 layers)
// ---------------------------------------------------------------------------
__global__ __launch_bounds__(256) void fo_pool_scan(
    const float* __restrict__ gates, float* __restrict__ xnext,
    float* __restrict__ csum) {
  int i = blockIdx.x * blockDim.x + threadIdx.x;   // 0 .. B*H-1
  if (i >= B_SZ * H_DIM) return;
  int b = i / H_DIM;
  int h = i - b * H_DIM;
  size_t base = (size_t)b * N3H + h;
  float c = 0.0f;
#pragma unroll 4
  for (int s = 0; s < S_LEN; ++s) {
    const float* g = gates + (size_t)s * (B_SZ * N3H) + base;
    float z = g[0];
    float f = g[H_DIM];
    float o = g[2 * H_DIM];
    c = f * c + (1.0f - f) * z;
    xnext[((size_t)s * B_SZ + b) * H_DIM + h] = o * c;
  }
  csum[i] += c;
}

__global__ void final_div(const float* __restrict__ csum,
                          const int* __restrict__ slen,
                          float* __restrict__ out) {
  int i = blockIdx.x * blockDim.x + threadIdx.x;
  if (i >= B_SZ * H_DIM) return;
  int b = i / H_DIM;
  out[i] = csum[i] / (float)slen[b];
}

// ---------------------------------------------------------------------------
// Host-side launch
// ---------------------------------------------------------------------------
extern "C" void kernel_launch(void* const* d_in, const int* in_sizes, int n_in,
                              void* d_out, int out_size, void* d_ws, size_t ws_size,
                              hipStream_t stream) {
  const float* sent  = (const float*)d_in[0];   // [S,B,E]
  const int*   slen  = (const int*)  d_in[1];   // [B]
  const float* W[3]  = { (const float*)d_in[2], (const float*)d_in[4], (const float*)d_in[6] };
  const float* bi[3] = { (const float*)d_in[3], (const float*)d_in[5], (const float*)d_in[7] };

  // Workspace layout (all offsets 16B aligned)
  char* ws = (char*)d_ws;
  const size_t GATES_B = (size_t)M_ROWS * N3H * sizeof(float);      // 235,929,600
  const size_t XBF_B   = (size_t)M_ROWS * KP_MAX * sizeof(__bf16);  //  39,845,888
  const size_t WBF_B   = (size_t)N3H * KP_MAX * sizeof(__bf16);     //  35,020,800
  const size_t XNXT_B  = (size_t)M_ROWS * H_DIM * sizeof(float);    //  78,643,200
  float*  gates = (float*) (ws);
  __bf16* xbf   = (__bf16*)(ws + GATES_B);
  __bf16* wbf   = (__bf16*)(ws + GATES_B + XBF_B);
  float*  xnext = (float*) (ws + GATES_B + XBF_B + WBF_B);
  float*  csum  = (float*) (ws + GATES_B + XBF_B + WBF_B + XNXT_B);

  const int BHN = B_SZ * H_DIM;                 // 76800
  dim3 gemmGrid(N3H / BN, M_ROWS / BM);         // (75, 64)

  zero_f32<<<(BHN + 255) / 256, 256, 0, stream>>>(csum, BHN);

  for (int layer = 0; layer < 3; ++layer) {
    const int K  = (layer == 0) ? E_DIM : H_DIM;
    const int Kp = (layer == 0) ? 320   : KP_MAX;  // pad K so KT = Kp/32 is even
    const float* xin = (layer == 0) ? sent : xnext;

    int nx = M_ROWS * Kp;
    cvt_pad_bf16<<<(nx + 255) / 256, 256, 0, stream>>>(xin, xbf, M_ROWS, K, Kp);
    int nw = N3H * Kp;
    cvt_pad_bf16<<<(nw + 255) / 256, 256, 0, stream>>>(W[layer], wbf, N3H, K, Kp);

    gemm_gates<<<gemmGrid, 256, 0, stream>>>(xbf, wbf, bi[layer], gates, Kp);

    fo_pool_scan<<<(BHN + 255) / 256, 256, 0, stream>>>(gates, xnext, csum);
  }

  final_div<<<(BHN + 255) / 256, 256, 0, stream>>>(csum, slen, (float*)d_out);
}
